// GGUFQuantizedLinear_86586540687542
// MI455X (gfx1250) — compile-verified, matrix operand order checked by reading, not run on hardware
//
#include <hip/hip_runtime.h>

typedef __attribute__((ext_vector_type(16))) _Float16 v16h;
typedef __attribute__((ext_vector_type(8)))  float    v8f;
typedef int i32x4 __attribute__((vector_size(16)));
typedef __attribute__((address_space(1))) i32x4* g4ptr_t;
typedef __attribute__((address_space(3))) i32x4* l4ptr_t;

#define BM 128
#define BN 128
#define BK 32
#define AP 40   // LDS row stride (halfs): 32 data + 8 pad = 80B (16B aligned)
#define BP 40

// ---------------------------------------------------------------------------
// Async global->LDS copy (CDNA5 GLOBAL_LOAD_ASYNC_TO_LDS_B128, ASYNCcnt)
// Builtin signature (from hipcc diagnostic): (i32x4 AS1*, i32x4 AS3*, Ii, Ii)
// ---------------------------------------------------------------------------
__device__ __forceinline__ void async_cp_b128(const _Float16* g, _Float16* l) {
#if __has_builtin(__builtin_amdgcn_global_load_async_to_lds_b128)
    __builtin_amdgcn_global_load_async_to_lds_b128(
        (g4ptr_t)(i32x4*)g,   // C-style: drop const, reinterpret, addrspacecast -> AS1
        (l4ptr_t)(i32x4*)l,   // addrspacecast generic -> AS3 (LDS)
        0, 0);
#else
    unsigned laddr = (unsigned)(uintptr_t)l;   // generic LDS addr: low 32 bits = LDS offset
    asm volatile("global_load_async_to_lds_b128 %0, %1, off"
                 :: "v"(laddr), "v"(g) : "memory");
#endif
}

__device__ __forceinline__ void wait_async0() {
#if __has_builtin(__builtin_amdgcn_s_wait_asynccnt)
    __builtin_amdgcn_s_wait_asynccnt(0);
#else
    asm volatile("s_wait_asynccnt 0x0" ::: "memory");
#endif
}

// ---------------------------------------------------------------------------
// Prep 1: x f32 [M][K] -> f16 [M][K]
// ---------------------------------------------------------------------------
__global__ __launch_bounds__(256)
void prep_x_kernel(const float* __restrict__ x, _Float16* __restrict__ x16) {
    const size_t idx = ((size_t)blockIdx.x * 256 + threadIdx.x) * 8;
    const float4* xp = reinterpret_cast<const float4*>(x + idx);
    float4 v0 = xp[0], v1 = xp[1];
    _Float16 h[8];
    h[0] = (_Float16)v0.x; h[1] = (_Float16)v0.y; h[2] = (_Float16)v0.z; h[3] = (_Float16)v0.w;
    h[4] = (_Float16)v1.x; h[5] = (_Float16)v1.y; h[6] = (_Float16)v1.z; h[7] = (_Float16)v1.w;
    *reinterpret_cast<uint4*>(x16 + idx) = *reinterpret_cast<const uint4*>(&h[0]);
}

// ---------------------------------------------------------------------------
// Prep 2: dequant qweight int32 [K][N] (+ blockscales) -> f16 TRANSPOSED [N][K]
// 128x128 tile per block, LDS transpose for coalesced writes.
// ---------------------------------------------------------------------------
#define TP 136   // LDS row stride (halfs) for 128-wide transpose tile (272B, 16B aligned)

__global__ __launch_bounds__(256)
void prep_w_kernel(const int* __restrict__ qw, const float* __restrict__ sc,
                   _Float16* __restrict__ wT, int N, int K) {
    __shared__ __align__(16) _Float16 tile[128 * TP];   // [n][k]

    const int t  = threadIdx.x;
    const int k0 = blockIdx.y * 128;
    const int n0 = blockIdx.x * 128;
    const int kc = (t >> 5) * 4;     // 0..28
    const int nc = (t & 31) * 4;     // 0..124

    #pragma unroll
    for (int it = 0; it < 4; ++it) {          // 4 passes of 32 k
        const int kb = it * 32;
        #pragma unroll
        for (int i = 0; i < 4; ++i) {
            const int kk = k0 + kb + kc + i;
            const int4 q = *reinterpret_cast<const int4*>(qw + (size_t)kk * N + n0 + nc);
            const float s = sc[(size_t)kk * (N >> 5) + ((n0 + nc) >> 5)];
            _Float16 h0 = (_Float16)((float)q.x * s);
            _Float16 h1 = (_Float16)((float)q.y * s);
            _Float16 h2 = (_Float16)((float)q.z * s);
            _Float16 h3 = (_Float16)((float)q.w * s);
            tile[(nc + 0) * TP + kb + kc + i] = h0;
            tile[(nc + 1) * TP + kb + kc + i] = h1;
            tile[(nc + 2) * TP + kb + kc + i] = h2;
            tile[(nc + 3) * TP + kb + kc + i] = h3;
        }
    }
    __syncthreads();

    // write rows of wT: row n = 128 contiguous f16 along k
    const int row  = t >> 1;
    const int part = (t & 1) * 64;
    const uint4* src = reinterpret_cast<const uint4*>(&tile[row * TP + part]);
    uint4* dst = reinterpret_cast<uint4*>(wT + (size_t)(n0 + row) * K + k0 + part);
    #pragma unroll
    for (int i = 0; i < 8; ++i) dst[i] = src[i];
}

// ---------------------------------------------------------------------------
// Main GEMM: f16 x f16^T via async-LDS double-buffered WMMA pipeline
// ---------------------------------------------------------------------------
__device__ __forceinline__ void compute_stage(const _Float16* __restrict__ sA,
                                              const _Float16* __restrict__ sB,
                                              v8f (&acc)[4][2],
                                              int wm, int wn, int l16, int lh) {
    union F { uint4 u[2]; v16h h; };
    F a[4], b[2];
    #pragma unroll
    for (int i = 0; i < 4; ++i) {
        const _Float16* p = &sA[(wm * 64 + i * 16 + l16) * AP + lh * 8];
        a[i].u[0] = *reinterpret_cast<const uint4*>(p);        // K 0-7 / 8-15
        a[i].u[1] = *reinterpret_cast<const uint4*>(p + 16);   // K 16-23 / 24-31
    }
    #pragma unroll
    for (int j = 0; j < 2; ++j) {
        const _Float16* p = &sB[(wn * 32 + j * 16 + l16) * BP + lh * 8];
        b[j].u[0] = *reinterpret_cast<const uint4*>(p);
        b[j].u[1] = *reinterpret_cast<const uint4*>(p + 16);
    }
    #pragma unroll
    for (int i = 0; i < 4; ++i)
        #pragma unroll
        for (int j = 0; j < 2; ++j)
            acc[i][j] = __builtin_amdgcn_wmma_f32_16x16x32_f16(
                false, a[i].h, false, b[j].h, (short)0, acc[i][j], false, false);
}

__global__ __launch_bounds__(256)
void gemm_f16_async_kernel(const _Float16* __restrict__ x16,
                           const _Float16* __restrict__ wT,
                           const float* __restrict__ bias,
                           float* __restrict__ out,
                           int M, int N, int K)
{
    __shared__ __align__(16) _Float16 sA0[BM * AP], sB0[BN * BP];
    __shared__ __align__(16) _Float16 sA1[BM * AP], sB1[BN * BP];

    const int t    = threadIdx.x;
    const int lane = t & 31;
    const int wave = t >> 5;
    const int wm   = wave & 1;
    const int wn   = wave >> 1;
    const int l16  = lane & 15;
    const int lh   = lane >> 4;

    const int m0 = blockIdx.y * BM;
    const int n0 = blockIdx.x * BN;

    // staging: 2 threads per 32-half row; each thread copies 16 halfs (2 x b128)
    const int row = t >> 1;
    const int off = (t & 1) * 16;

    const _Float16* gA = x16 + (size_t)(m0 + row) * K + off;   // + k0
    const _Float16* gB = wT  + (size_t)(n0 + row) * K + off;   // + k0
    _Float16* lA0 = &sA0[row * AP + off];
    _Float16* lB0 = &sB0[row * BP + off];
    _Float16* lA1 = &sA1[row * AP + off];
    _Float16* lB1 = &sB1[row * BP + off];

    v8f acc[4][2] = {};
    const int nK = K / BK;   // 128 (even)

    // Prologue: async tile 0 -> buf0
    async_cp_b128(gA, lA0);     async_cp_b128(gA + 8, lA0 + 8);
    async_cp_b128(gB, lB0);     async_cp_b128(gB + 8, lB0 + 8);
    wait_async0();
    __syncthreads();

    for (int kt = 0; kt < nK; kt += 2) {
        // ---- even stage: compute buf0 (tile kt), async tile kt+1 -> buf1 ----
        {
            const int k1 = (kt + 1 < nK) ? (kt + 1) * BK : 0;   // clamped: dead on last iter
            async_cp_b128(gA + k1, lA1);     async_cp_b128(gA + k1 + 8, lA1 + 8);
            async_cp_b128(gB + k1, lB1);     async_cp_b128(gB + k1 + 8, lB1 + 8);

            const int kpf = (kt + 3 < nK ? kt + 3 : 0) * BK;
            __builtin_prefetch(gA + kpf, 0, 3);
            __builtin_prefetch(gB + kpf, 0, 3);

            compute_stage(sA0, sB0, acc, wm, wn, l16, lh);
            wait_async0();
        }
        __syncthreads();

        // ---- odd stage: compute buf1 (tile kt+1), async tile kt+2 -> buf0 ----
        {
            const int k2 = (kt + 2 < nK) ? (kt + 2) * BK : 0;
            async_cp_b128(gA + k2, lA0);     async_cp_b128(gA + k2 + 8, lA0 + 8);
            async_cp_b128(gB + k2, lB0);     async_cp_b128(gB + k2 + 8, lB0 + 8);

            const int kpf = (kt + 4 < nK ? kt + 4 : 0) * BK;
            __builtin_prefetch(gA + kpf, 0, 3);
            __builtin_prefetch(gB + kpf, 0, 3);

            compute_stage(sA1, sB1, acc, wm, wn, l16, lh);
            wait_async0();
        }
        __syncthreads();
    }

    // ---- Epilogue: C/D layout (lane L: N = L%16; VGPR v: M = (L/16)*8 + v) + bias ----
    #pragma unroll
    for (int j = 0; j < 2; ++j) {
        const int col = n0 + wn * 32 + j * 16 + l16;
        const float bv = bias[col];
        #pragma unroll
        for (int i = 0; i < 4; ++i) {
            const int rbase = m0 + wm * 64 + i * 16 + lh * 8;
            #pragma unroll
            for (int v = 0; v < 8; ++v)
                out[(size_t)(rbase + v) * N + col] = acc[i][j][v] + bv;
        }
    }
}

// ---------------------------------------------------------------------------
// Fallback (no workspace): fused dequant GEMM
// ---------------------------------------------------------------------------
struct AStage { float4 v[4]; };
struct BStage { int4 q[4]; float s[4]; };

__device__ __forceinline__ void load_A(const float* __restrict__ x, size_t arowbase,
                                       int k0, AStage& a) {
    const float4* xp = reinterpret_cast<const float4*>(x + arowbase + k0);
    #pragma unroll
    for (int i = 0; i < 4; ++i) a.v[i] = xp[i];
}

__device__ __forceinline__ void load_B(const int* __restrict__ qw, const float* __restrict__ sc,
                                       int k0, int kc, int col, int N, BStage& b) {
    #pragma unroll
    for (int i = 0; i < 4; ++i) {
        const int kk = k0 + kc + i;
        b.q[i] = *reinterpret_cast<const int4*>(qw + (size_t)kk * N + col);
        b.s[i] = sc[(size_t)kk * (N >> 5) + (col >> 5)];
    }
}

__device__ __forceinline__ void store_stage(_Float16* __restrict__ sA, _Float16* __restrict__ sB,
                                            const AStage& a, const BStage& b,
                                            int arow, int ahalf, int kc, int nc) {
    _Float16 h[16];
    #pragma unroll
    for (int i = 0; i < 4; ++i) {
        h[i * 4 + 0] = (_Float16)a.v[i].x;
        h[i * 4 + 1] = (_Float16)a.v[i].y;
        h[i * 4 + 2] = (_Float16)a.v[i].z;
        h[i * 4 + 3] = (_Float16)a.v[i].w;
    }
    uint4* dst = reinterpret_cast<uint4*>(&sA[arow * AP + ahalf]);
    dst[0] = *reinterpret_cast<const uint4*>(&h[0]);
    dst[1] = *reinterpret_cast<const uint4*>(&h[8]);

    _Float16 hb[4][4];
    #pragma unroll
    for (int i = 0; i < 4; ++i) {
        const float s = b.s[i];
        hb[0][i] = (_Float16)((float)b.q[i].x * s);
        hb[1][i] = (_Float16)((float)b.q[i].y * s);
        hb[2][i] = (_Float16)((float)b.q[i].z * s);
        hb[3][i] = (_Float16)((float)b.q[i].w * s);
    }
    #pragma unroll
    for (int j = 0; j < 4; ++j)
        *reinterpret_cast<uint2*>(&sB[(nc + j) * BP + kc]) =
            *reinterpret_cast<const uint2*>(&hb[j][0]);
}

__global__ __launch_bounds__(256)
void gemm_fused_fallback_kernel(const float* __restrict__ x,
                                const int*   __restrict__ qw,
                                const float* __restrict__ sc,
                                const float* __restrict__ bias,
                                float* __restrict__ out,
                                int M, int N, int K)
{
    __shared__ __align__(16) _Float16 sA0[BM * AP], sB0[BN * BP];
    __shared__ __align__(16) _Float16 sA1[BM * AP], sB1[BN * BP];

    const int t    = threadIdx.x;
    const int lane = t & 31;
    const int wave = t >> 5;
    const int wm   = wave & 1;
    const int wn   = wave >> 1;
    const int l16  = lane & 15;
    const int lh   = lane >> 4;

    const int m0 = blockIdx.y * BM;
    const int n0 = blockIdx.x * BN;

    const int arow  = t >> 1;
    const int ahalf = (t & 1) * 16;
    const int kc    = wave * 4;
    const int nc    = lane * 4;

    const size_t arowbase = (size_t)(m0 + arow) * K + ahalf;
    const int    bcol     = n0 + nc;

    v8f acc[4][2] = {};
    const int nK = K / BK;

    {
        AStage a0; BStage b0;
        load_A(x, arowbase, 0, a0);
        load_B(qw, sc, 0, kc, bcol, N, b0);
        store_stage(sA0, sB0, a0, b0, arow, ahalf, kc, nc);
    }
    __syncthreads();

    for (int kt = 0; kt < nK; kt += 2) {
        {
            const int k1 = (kt + 1 < nK) ? (kt + 1) * BK : 0;
            AStage a1; BStage b1;
            load_A(x, arowbase, k1, a1);
            load_B(qw, sc, k1, kc, bcol, N, b1);
            compute_stage(sA0, sB0, acc, wm, wn, l16, lh);
            store_stage(sA1, sB1, a1, b1, arow, ahalf, kc, nc);
        }
        __syncthreads();
        {
            const int k2 = (kt + 2 < nK) ? (kt + 2) * BK : 0;
            AStage a2; BStage b2;
            load_A(x, arowbase, k2, a2);
            load_B(qw, sc, k2, kc, bcol, N, b2);
            compute_stage(sA1, sB1, acc, wm, wn, l16, lh);
            store_stage(sA0, sB0, a2, b2, arow, ahalf, kc, nc);
        }
        __syncthreads();
    }

    #pragma unroll
    for (int j = 0; j < 2; ++j) {
        const int col = n0 + wn * 32 + j * 16 + l16;
        const float bv = bias[col];
        #pragma unroll
        for (int i = 0; i < 4; ++i) {
            const int rbase = m0 + wm * 64 + i * 16 + lh * 8;
            #pragma unroll
            for (int v = 0; v < 8; ++v)
                out[(size_t)(rbase + v) * N + col] = acc[i][j][v] + bv;
        }
    }
}

// ---------------------------------------------------------------------------
extern "C" void kernel_launch(void* const* d_in, const int* in_sizes, int n_in,
                              void* d_out, int out_size, void* d_ws, size_t ws_size,
                              hipStream_t stream) {
    const float* x    = (const float*)d_in[0];   // [B,S,IN] f32
    const int*   qw   = (const int*)d_in[1];     // [IN,OUT] int8 codes as int32
    const float* sc   = (const float*)d_in[2];   // [IN,OUT/32] f32
    const float* bias = (const float*)d_in[3];   // [OUT] f32
    float* out = (float*)d_out;                  // [B,S,OUT] f32

    const int K = 4096;
    const int N = 4096;
    const int M = in_sizes[0] / K;               // 8192

    const size_t need = ((size_t)M * K + (size_t)N * K) * sizeof(_Float16);  // 96 MB
    if (ws_size >= need) {
        _Float16* x16 = (_Float16*)d_ws;                    // [M][K]
        _Float16* wT  = x16 + (size_t)M * K;                // [N][K]
        prep_x_kernel<<<(int)(((size_t)M * K) / (256 * 8)), 256, 0, stream>>>(x, x16);
        prep_w_kernel<<<dim3(N / 128, K / 128), 256, 0, stream>>>(qw, sc, wT, N, K);
        gemm_f16_async_kernel<<<dim3(N / BN, M / BM), 256, 0, stream>>>(x16, wT, bias, out, M, N, K);
    } else {
        gemm_fused_fallback_kernel<<<dim3(N / BN, M / BM), 256, 0, stream>>>(x, qw, sc, bias, out, M, N, K);
    }
}
